// MFU_module_22754736734974
// MI455X (gfx1250) — matrix operand order repaired and checked by reading, not compiled, not run on hardware
//
#include <hip/hip_runtime.h>
#include <math.h>

// ---------------- constants ----------------
#define HWPIX 4096          // 64*64
#define NTOT  8192          // B(2) * HWPIX

typedef __attribute__((ext_vector_type(16))) __bf16 v16bf;
typedef __attribute__((ext_vector_type(8)))  float  v8f;

union BfFrag { uint4 u[2]; v16bf v; };

__device__ __forceinline__ unsigned short mfu_f2bf(float f) {
  unsigned u = __float_as_uint(f);
  u += 0x7FFFu + ((u >> 16) & 1u);           // round-to-nearest-even
  return (unsigned short)(u >> 16);
}

__device__ __forceinline__ int mfu_reflect(int i) {
  i = (i < 0) ? -i : i;
  return (i > 63) ? (126 - i) : i;
}

// ---------------- WMMA GEMM: D[M,8192] = A[M,K](bf16) x Bt[8192,K]^T (bf16) ----
// A row-major [Mpad][K] (Mpad multiple of 32), Bt row-major [N][K] (B col-major).
// One wave per 32x32 output block = 2x2 WMMA tiles with A/B fragment reuse:
// 8 b128 loads feed 4 v_wmma (2x the wmma-per-load density of 1 tile/wave).
// K multiple of 32. Output f32 row-major, stride 8192.
__global__ __launch_bounds__(32) void k_gemm(
    const unsigned short* __restrict__ A, const unsigned short* __restrict__ Bt,
    float* __restrict__ D, const float* __restrict__ bias,
    int K, int Mreal, int relu) {
  const int lane = threadIdx.x;
  const int half = lane >> 4;
  const int l16  = lane & 15;
  const int ntile = blockIdx.x << 5;   // 32 columns
  const int mtile = blockIdx.y << 5;   // 32 rows
  const unsigned short* a0 = A  + (size_t)(mtile + l16) * K;
  const unsigned short* a1 = A  + (size_t)(mtile + 16 + l16) * K;
  const unsigned short* b0 = Bt + (size_t)(ntile + l16) * K;
  const unsigned short* b1 = Bt + (size_t)(ntile + 16 + l16) * K;
  v8f acc00 = {}, acc01 = {}, acc10 = {}, acc11 = {};
  for (int k0 = 0; k0 < K; k0 += 32) {
    BfFrag af0, af1, bf0, bf1;
    // A 16x32 bf16 layout: lanes0-15 K{0..7,16..23}, lanes16-31 K{8..15,24..31}
    af0.u[0] = *(const uint4*)(a0 + k0 + 8 * half);
    af0.u[1] = *(const uint4*)(a0 + k0 + 16 + 8 * half);
    af1.u[0] = *(const uint4*)(a1 + k0 + 8 * half);
    af1.u[1] = *(const uint4*)(a1 + k0 + 16 + 8 * half);
    // B 32x16 bf16 layout: lanes0-15 K{0..7,16..23}, lanes16-31 K{8..15,24..31} (col = lane%16)
    bf0.u[0] = *(const uint4*)(b0 + k0 + 8 * half);
    bf0.u[1] = *(const uint4*)(b0 + k0 + 16 + 8 * half);
    bf1.u[0] = *(const uint4*)(b1 + k0 + 8 * half);
    bf1.u[1] = *(const uint4*)(b1 + k0 + 16 + 8 * half);
    acc00 = __builtin_amdgcn_wmma_f32_16x16x32_bf16(false, af0.v, false, bf0.v,
                                                    (short)0, acc00, false, false);
    acc01 = __builtin_amdgcn_wmma_f32_16x16x32_bf16(false, af0.v, false, bf1.v,
                                                    (short)0, acc01, false, false);
    acc10 = __builtin_amdgcn_wmma_f32_16x16x32_bf16(false, af1.v, false, bf0.v,
                                                    (short)0, acc10, false, false);
    acc11 = __builtin_amdgcn_wmma_f32_16x16x32_bf16(false, af1.v, false, bf1.v,
                                                    (short)0, acc11, false, false);
  }
  // C/D layout: VGPR r -> M = r + 8*(lane/16), N = lane%16
#pragma unroll
  for (int t = 0; t < 4; ++t) {
    const v8f& acc = (t == 0) ? acc00 : (t == 1) ? acc01 : (t == 2) ? acc10 : acc11;
    int mbase = mtile + ((t >> 1) << 4);
    int nbase = ntile + ((t & 1) << 4);
#pragma unroll
    for (int r = 0; r < 8; ++r) {
      int mr = mbase + r + 8 * half;
      float val = acc[r];
      if (bias && mr < Mreal) val += bias[mr];
      if (relu) val = fmaxf(val, 0.f);
      D[(size_t)mr * NTOT + nbase + l16] = val;
    }
  }
}

// ---------------- weight f32 -> bf16 with M/K padding ----------------
__global__ __launch_bounds__(256) void k_wconv(const float* __restrict__ w,
    unsigned short* __restrict__ out, int M, int K, int Kpad, int total) {
  int idx = blockIdx.x * 256 + threadIdx.x;
  if (idx >= total) return;
  int m = idx / Kpad, k = idx - m * Kpad;
  float v = (m < M && k < K) ? w[(size_t)m * K + k] : 0.f;
  out[idx] = mfu_f2bf(v);
}

// ---------------- activation -> [N][Kpad] bf16 (Bt) ----------------
// in value at in[c*cstride + b*bstride + p], n = b*4096 + p.
// span==Kpad : write whole row (zero outside [kofs,kofs+C)); else write C cols.
__global__ __launch_bounds__(256) void k_act2nk(const float* __restrict__ in,
    unsigned short* __restrict__ out, long long cstride, long long bstride,
    int C, int Kpad, int kofs, int span, int total) {
  int idx = blockIdx.x * 256 + threadIdx.x;
  if (idx >= total) return;
  int n = idx / span, j = idx - n * span;
  int b = n >> 12, p = n & 4095;
  int k, c;
  if (span == Kpad) { k = j; c = j - kofs; }
  else             { k = kofs + j; c = j; }
  float v = 0.f;
  if (c >= 0 && c < C) v = in[(long long)c * cstride + (long long)b * bstride + p];
  out[(size_t)n * Kpad + k] = mfu_f2bf(v);
}

// ---------------- im2col 3x3 pad=1 -> [8192][576] bf16 ----------------
__global__ __launch_bounds__(256) void k_im2col(const float* __restrict__ in,
    unsigned short* __restrict__ out, long long cstride, long long bstride) {
  int idx = blockIdx.x * 256 + threadIdx.x;       // < 8192*576
  int n = idx / 576, k = idx - n * 576;
  int c = k / 9, r = k - c * 9;
  int ky = r / 3, kx = r - ky * 3;
  int b = n >> 12, p = n & 4095, y = p >> 6, x = p & 63;
  int sy = y + ky - 1, sx = x + kx - 1;
  float v = 0.f;
  if ((unsigned)sy < 64u && (unsigned)sx < 64u)
    v = in[(long long)c * cstride + (long long)b * bstride + sy * 64 + sx];
  out[idx] = mfu_f2bf(v);
}

// ---------------- BatchNorm (training-mode batch stats) ----------------
__global__ __launch_bounds__(256) void k_bnstats(const float* __restrict__ x,
                                                 float* __restrict__ stats) {
  int c = blockIdx.x;
  __shared__ float ss[256], sq[256];
  float s = 0.f, q = 0.f;
  for (int i = threadIdx.x; i < NTOT; i += 256) {
    float v = x[(size_t)c * NTOT + i];
    s += v; q += v * v;
  }
  ss[threadIdx.x] = s; sq[threadIdx.x] = q;
  __syncthreads();
  for (int o = 128; o > 0; o >>= 1) {
    if (threadIdx.x < o) { ss[threadIdx.x] += ss[threadIdx.x + o];
                           sq[threadIdx.x] += sq[threadIdx.x + o]; }
    __syncthreads();
  }
  if (threadIdx.x == 0) {
    float m = ss[0] * (1.f / NTOT);
    float var = sq[0] * (1.f / NTOT) - m * m;
    stats[2 * c] = m;
    stats[2 * c + 1] = rsqrtf(var + 1e-5f);
  }
}

__global__ __launch_bounds__(256) void k_bnapply(const float* __restrict__ x,
    float* __restrict__ y, const float* __restrict__ stats,
    const float* __restrict__ g, const float* __restrict__ beta,
    int relu, int total) {
  int idx = blockIdx.x * 256 + threadIdx.x;
  if (idx >= total) return;
  int c = idx >> 13;
  float v = (x[idx] - stats[2 * c]) * stats[2 * c + 1] * g[c] + beta[c];
  if (relu) v = fmaxf(v, 0.f);
  y[idx] = v;
}

// ---------------- ACmix pieces ----------------
__global__ __launch_bounds__(256) void k_pe(const float* __restrict__ wp,
    const float* __restrict__ bp, float* __restrict__ pe) {
  int idx = blockIdx.x * 256 + threadIdx.x;       // 16*4096
  int d = idx >> 12, p = idx & 4095;
  float lw = -1.f + (2.f / 63.f) * (float)(p & 63);
  float lh = -1.f + (2.f / 63.f) * (float)(p >> 6);
  pe[idx] = wp[2 * d] * lw + wp[2 * d + 1] * lh + bp[d];
}

__global__ __launch_bounds__(256) void k_attn(const float* __restrict__ q,
    const float* __restrict__ k, const float* __restrict__ v,
    const float* __restrict__ pe, float* __restrict__ out) {
  int idx = blockIdx.x * 256 + threadIdx.x;       // 8 * 4096
  int bh = idx >> 12, p = idx & 4095;
  int b = bh >> 2, c0 = (bh & 3) << 4;
  int y = p >> 6, x = p & 63;
  size_t nb = (size_t)b * HWPIX;
  float qv[16], pc[16];
#pragma unroll
  for (int d = 0; d < 16; ++d) {
    qv[d] = q[(size_t)(c0 + d) * NTOT + nb + p] * 0.25f;   // HD^-0.5
    pc[d] = pe[d * HWPIX + p];
  }
  float att[49];
  float mx = -1e30f;
  for (int r = 0; r < 49; ++r) {
    int sy = mfu_reflect(y + r / 7 - 3);
    int sx = mfu_reflect(x + r % 7 - 3);
    int np = sy * 64 + sx;
    float s = 0.f;
#pragma unroll
    for (int d = 0; d < 16; ++d)
      s += qv[d] * (k[(size_t)(c0 + d) * NTOT + nb + np] + pc[d] - pe[d * HWPIX + np]);
    att[r] = s;
    mx = fmaxf(mx, s);
  }
  float sum = 0.f;
  for (int r = 0; r < 49; ++r) { att[r] = __expf(att[r] - mx); sum += att[r]; }
  float inv = 1.f / sum;
  float o[16];
#pragma unroll
  for (int d = 0; d < 16; ++d) o[d] = 0.f;
  for (int r = 0; r < 49; ++r) {
    int sy = mfu_reflect(y + r / 7 - 3);
    int sx = mfu_reflect(x + r % 7 - 3);
    int np = sy * 64 + sx;
    float w = att[r] * inv;
#pragma unroll
    for (int d = 0; d < 16; ++d)
      o[d] += w * v[(size_t)(c0 + d) * NTOT + nb + np];
  }
#pragma unroll
  for (int d = 0; d < 16; ++d)
    out[(size_t)(c0 + d) * NTOT + nb + p] = o[d];
}

// f_conv[d*9+o][n] = sum_i wfc[o,i] * qkv_i[d][n]
__global__ __launch_bounds__(256) void k_fconv(const float* __restrict__ q,
    const float* __restrict__ k, const float* __restrict__ v,
    const float* __restrict__ wfc, float* __restrict__ fc) {
  int idx = blockIdx.x * 256 + threadIdx.x;       // 144 * 8192
  int row = idx >> 13, n = idx & 8191;
  int d = row / 9, o = row - d * 9;
  float s = 0.f;
#pragma unroll
  for (int h = 0; h < 4; ++h) {
    int c = (h << 4) + d;
    s += wfc[o * 12 + h]     * q[(size_t)c * NTOT + n];
    s += wfc[o * 12 + 4 + h] * k[(size_t)c * NTOT + n];
    s += wfc[o * 12 + 8 + h] * v[(size_t)c * NTOT + n];
  }
  fc[idx] = s;
}

// grouped dep_conv 3x3 pad=1, groups=16 (in 9/group, out 4/group), no bias
__global__ __launch_bounds__(256) void k_depconv(const float* __restrict__ fc,
    const float* __restrict__ wd, float* __restrict__ oc) {
  int idx = blockIdx.x * 256 + threadIdx.x;       // 64 * 8192
  int c = idx >> 13, n = idx & 8191;
  int b = n >> 12, p = n & 4095, y = p >> 6, x = p & 63;
  int d = c >> 2;
  float s = 0.f;
  for (int o = 0; o < 9; ++o) {
    const float* fr = fc + (size_t)(d * 9 + o) * NTOT + (size_t)b * HWPIX;
#pragma unroll
    for (int ky = 0; ky < 3; ++ky) {
      int sy = y + ky - 1; if ((unsigned)sy > 63u) continue;
#pragma unroll
      for (int kx = 0; kx < 3; ++kx) {
        int sx = x + kx - 1; if ((unsigned)sx > 63u) continue;
        s += wd[c * 81 + o * 9 + ky * 3 + kx] * fr[sy * 64 + sx];
      }
    }
  }
  oc[idx] = s;
}

// ---------------- elementwise ----------------
__global__ __launch_bounds__(256) void k_axpby(const float* __restrict__ a,
    const float* __restrict__ b, const float* __restrict__ r1,
    const float* __restrict__ r2, float* __restrict__ o, int n) {
  int i = blockIdx.x * 256 + threadIdx.x;
  if (i < n) o[i] = r1[0] * a[i] + r2[0] * b[i];
}

__global__ __launch_bounds__(256) void k_fsum(const float* __restrict__ sa,
    const float* __restrict__ sc, const float* __restrict__ enc,
    float* __restrict__ o, int n) {
  int i = blockIdx.x * 256 + threadIdx.x;
  if (i < n) o[i] = (sa[i] + sc[i]) * (1.f / (1.f + __expf(-enc[i])));
}

__global__ __launch_bounds__(256) void k_add(const float* __restrict__ a,
    const float* __restrict__ b, float* __restrict__ o, int n) {
  int i = blockIdx.x * 256 + threadIdx.x;
  if (i < n) o[i] = a[i] + b[i];
}

__global__ __launch_bounds__(256) void k_mix(const float* __restrict__ xv,
    const float* __restrict__ res, const float* __restrict__ l,
    const float* __restrict__ g, float* __restrict__ o, int n) {
  int i = blockIdx.x * 256 + threadIdx.x;
  if (i < n) {
    float w = 1.f / (1.f + __expf(-(l[i] + g[i])));
    o[i] = xv[i] * w + res[i] * (1.f - w);
  }
}

// FIN[c][b*4096+p] + x(NCHW) -> out(NCHW)
__global__ __launch_bounds__(256) void k_final(const float* __restrict__ fin,
    const float* __restrict__ x, float* __restrict__ out) {
  int idx = blockIdx.x * 256 + threadIdx.x;       // 2*512*4096
  int b = idx >> 21;
  int rem = idx & 2097151;
  int c = rem >> 12, p = rem & 4095;
  out[idx] = fin[(size_t)c * NTOT + (size_t)b * HWPIX + p] + x[idx];
}

// ================= host orchestration =================
struct Acx { const float *wq,*bq,*wk,*bk,*wv,*bv,*wp,*bp,*wfc,*wdep,*r1,*r2; };
struct Dan { const float *c5aw,*c5ag,*c5ab,*c5cw,*c5cg,*c5cb;
             Acx sa, sc;
             const float *c51w,*c51g,*c51b,*c52w,*c52g,*c52b,*c8w,*c8b; };
struct Br  { const float *w1,*b1,*g1,*bt1,*w2,*b2,*g2,*bt2; };

extern "C" void kernel_launch(void* const* d_in, const int* in_sizes, int n_in,
                              void* d_out, int out_size, void* d_ws, size_t ws_size,
                              hipStream_t stream) {
  (void)in_sizes; (void)n_in; (void)out_size; (void)ws_size;

  // ---- read inputs in setup_inputs() dict order ----
  int ip = 0;
  auto P = [&]() { return (const float*)d_in[ip++]; };
  const float* x_in    = P();
  const float* d_prior = P();
  const float* re_w1 = P(); const float* re_b1 = P();
  const float* re_w2 = P(); const float* re_b2 = P();
  auto readAcx = [&](Acx& a) {
    a.wq = P(); a.bq = P(); a.wk = P(); a.bk = P(); a.wv = P(); a.bv = P();
    a.wp = P(); a.bp = P(); a.wfc = P(); a.wdep = P(); a.r1 = P(); a.r2 = P();
  };
  Dan dan[6];
  for (int i = 0; i < 6; ++i) {
    Dan& d = dan[i];
    d.c5aw = P(); d.c5ag = P(); d.c5ab = P();
    d.c5cw = P(); d.c5cg = P(); d.c5cb = P();
    readAcx(d.sa); readAcx(d.sc);
    d.c51w = P(); d.c51g = P(); d.c51b = P();
    d.c52w = P(); d.c52g = P(); d.c52b = P();
    d.c8w = P(); d.c8b = P();
  }
  auto readBr = [&](Br& b) {
    b.w1 = P(); b.b1 = P(); b.g1 = P(); b.bt1 = P();
    b.w2 = P(); b.b2 = P(); b.g2 = P(); b.bt2 = P();
  };
  Br brL, brG, brL2;
  readBr(brL); readBr(brG); readBr(brL2);
  const float* fin_w = P(); const float* fin_b = P();

  // ---- workspace bump allocator ----
  char* wsp = (char*)d_ws;
  size_t off = 0;
  auto alloc = [&](size_t bytes) -> void* {
    void* p = wsp + off;
    off = (off + bytes + 255) & ~(size_t)255;
    return p;
  };
  float* ENC = (float*)alloc(512ull * NTOT * 4);   // reused as FIN at the end
  float* FEAT[6]; for (int i = 0; i < 6; ++i) FEAT[i] = (float*)alloc(64ull * NTOT * 4);
  float* A123[3]; for (int i = 0; i < 3; ++i) A123[i] = (float*)alloc(64ull * NTOT * 4);
  unsigned short* COLX = (unsigned short*)alloc(8192ull * 576 * 2);
  unsigned short* COLM = (unsigned short*)alloc(8192ull * 576 * 2);
  unsigned short* BT   = (unsigned short*)alloc(8192ull * 512 * 2);
  float *T0 = (float*)alloc(64ull*NTOT*4), *F1 = (float*)alloc(64ull*NTOT*4);
  float *F2 = (float*)alloc(64ull*NTOT*4), *Q  = (float*)alloc(64ull*NTOT*4);
  float *Kb = (float*)alloc(64ull*NTOT*4), *V  = (float*)alloc(64ull*NTOT*4);
  float *OA = (float*)alloc(64ull*NTOT*4), *OC = (float*)alloc(64ull*NTOT*4);
  float *ACM= (float*)alloc(64ull*NTOT*4), *SA = (float*)alloc(64ull*NTOT*4);
  float *SC = (float*)alloc(64ull*NTOT*4), *FS = (float*)alloc(64ull*NTOT*4);
  float* FC = (float*)alloc(144ull * NTOT * 4);
  float* PE = (float*)alloc(16ull * HWPIX * 4);
  float* Hb = (float*)alloc(96ull * NTOT * 4);     // enc hidden (96 pad rows) / iaff hidden (32 pad rows)
  float* STATS = (float*)alloc(4096);
  unsigned short *wbC[6][4], *wbQ[6][2], *wbK[6][2], *wbV[6][2], *wbC8[6];
  for (int i = 0; i < 6; ++i) {
    for (int j = 0; j < 4; ++j) wbC[i][j] = (unsigned short*)alloc(64ull * 576 * 2);
    for (int j = 0; j < 2; ++j) {
      wbQ[i][j] = (unsigned short*)alloc(64ull * 64 * 2);
      wbK[i][j] = (unsigned short*)alloc(64ull * 64 * 2);
      wbV[i][j] = (unsigned short*)alloc(64ull * 64 * 2);
    }
    wbC8[i] = (unsigned short*)alloc(64ull * 64 * 2);
  }
  unsigned short *wbI1[3], *wbI2[3];
  for (int j = 0; j < 3; ++j) {
    wbI1[j] = (unsigned short*)alloc(32ull * 64 * 2);   // Mpad 16 -> 32
    wbI2[j] = (unsigned short*)alloc(64ull * 32 * 2);
  }
  unsigned short* wbRE1 = (unsigned short*)alloc(96ull * 32 * 2);   // Mpad 72 -> 96
  unsigned short* wbRE2 = (unsigned short*)alloc(512ull * 96 * 2);
  unsigned short* wbFIN = (unsigned short*)alloc(512ull * 512 * 2);

  // ---- launch helpers ----
  auto convw = [&](const float* w, unsigned short* dst, int M, int K, int Mpad, int Kpad) {
    int tot = Mpad * Kpad;
    k_wconv<<<(tot + 255) / 256, 256, 0, stream>>>(w, dst, M, K, Kpad, tot);
  };
  // Mt32 = Mpad/32 (rows per 32-row wave block)
  auto gemm = [&](const unsigned short* A, const unsigned short* Bt, float* D,
                  const float* bias, int Mt32, int K, int Mreal, int relu) {
    dim3 g(NTOT / 32, Mt32);
    k_gemm<<<g, 32, 0, stream>>>(A, Bt, D, bias, K, Mreal, relu);
  };
  auto a2nk = [&](const float* in, long long cs, long long bs, int C, int Kpad,
                  int kofs, bool full) {
    int span = full ? Kpad : C;
    int tot = NTOT * span;
    k_act2nk<<<(tot + 255) / 256, 256, 0, stream>>>(in, BT, cs, bs, C, Kpad,
                                                    kofs, span, tot);
  };
  auto im2col = [&](const float* in, long long cs, long long bs, unsigned short* col) {
    k_im2col<<<(8192 * 576) / 256, 256, 0, stream>>>(in, col, cs, bs);
  };
  auto bn = [&](const float* src, float* dst, const float* g_, const float* b_,
                int C, int relu) {
    k_bnstats<<<C, 256, 0, stream>>>(src, STATS);
    int tot = C * NTOT;
    k_bnapply<<<tot / 256, 256, 0, stream>>>(src, dst, STATS, g_, b_, relu, tot);
  };
  const int CN = 64 * NTOT;                 // 524288
  auto acmix = [&](const float* feat, const Acx& p, const unsigned short* wq,
                   const unsigned short* wk, const unsigned short* wv, float* out) {
    a2nk(feat, NTOT, HWPIX, 64, 64, 0, true);
    gemm(wq, BT, Q,  p.bq, 2, 64, 64, 0);
    gemm(wk, BT, Kb, p.bk, 2, 64, 64, 0);
    gemm(wv, BT, V,  p.bv, 2, 64, 64, 0);
    k_pe<<<(16 * HWPIX) / 256, 256, 0, stream>>>(p.wp, p.bp, PE);
    k_attn<<<(8 * HWPIX) / 256, 256, 0, stream>>>(Q, Kb, V, PE, OA);
    k_fconv<<<(144 * NTOT) / 256, 256, 0, stream>>>(Q, Kb, V, p.wfc, FC);
    k_depconv<<<CN / 256, 256, 0, stream>>>(FC, p.wdep, OC);
    k_axpby<<<CN / 256, 256, 0, stream>>>(OA, OC, p.r1, p.r2, out, CN);
  };

  // ---- convert all weights to bf16 (padded) ----
  for (int i = 0; i < 6; ++i) {
    Dan& d = dan[i];
    convw(d.c5aw, wbC[i][0], 64, 576, 64, 576);
    convw(d.c5cw, wbC[i][1], 64, 576, 64, 576);
    convw(d.c51w, wbC[i][2], 64, 576, 64, 576);
    convw(d.c52w, wbC[i][3], 64, 576, 64, 576);
    convw(d.sa.wq, wbQ[i][0], 64, 64, 64, 64);
    convw(d.sa.wk, wbK[i][0], 64, 64, 64, 64);
    convw(d.sa.wv, wbV[i][0], 64, 64, 64, 64);
    convw(d.sc.wq, wbQ[i][1], 64, 64, 64, 64);
    convw(d.sc.wk, wbK[i][1], 64, 64, 64, 64);
    convw(d.sc.wv, wbV[i][1], 64, 64, 64, 64);
    convw(d.c8w, wbC8[i], 64, 64, 64, 64);
  }
  const Br* brs[3] = {&brL, &brG, &brL2};
  for (int j = 0; j < 3; ++j) {
    convw(brs[j]->w1, wbI1[j], 16, 64, 32, 64);
    convw(brs[j]->w2, wbI2[j], 64, 16, 64, 32);
  }
  convw(re_w1, wbRE1, 72, 9, 96, 32);
  convw(re_w2, wbRE2, 512, 72, 512, 96);
  convw(fin_w, wbFIN, 512, 512, 512, 512);

  // ---- enc = conv1x1(relu(conv1x1(d_prior))) ----
  a2nk(d_prior, HWPIX, 9ll * HWPIX, 9, 32, 0, true);      // NCHW [2,9,64,64]
  gemm(wbRE1, BT, Hb, re_b1, 3, 32, 72, 1);
  a2nk(Hb, NTOT, HWPIX, 72, 96, 0, true);                 // CN layout
  gemm(wbRE2, BT, ENC, re_b2, 16, 96, 512, 0);

  // ---- 6 DANet heads ----
  for (int i = 0; i < 6; ++i) {
    Dan& d = dan[i];
    const float* xs = x_in + (size_t)i * 64 * HWPIX;      // NCHW slice base
    im2col(xs, HWPIX, 512ll * HWPIX, COLX);
    // branch a
    gemm(wbC[i][0], COLX, T0, nullptr, 2, 576, 64, 0);
    bn(T0, F1, d.c5ag, d.c5ab, 64, 1);
    acmix(F1, d.sa, wbQ[i][0], wbK[i][0], wbV[i][0], ACM);
    im2col(ACM, NTOT, HWPIX, COLM);
    gemm(wbC[i][2], COLM, T0, nullptr, 2, 576, 64, 0);
    bn(T0, SA, d.c51g, d.c51b, 64, 1);
    // branch c
    gemm(wbC[i][1], COLX, T0, nullptr, 2, 576, 64, 0);
    bn(T0, F2, d.c5cg, d.c5cb, 64, 1);
    acmix(F2, d.sc, wbQ[i][1], wbK[i][1], wbV[i][1], ACM);
    im2col(ACM, NTOT, HWPIX, COLM);
    gemm(wbC[i][3], COLM, T0, nullptr, 2, 576, 64, 0);
    bn(T0, SC, d.c52g, d.c52b, 64, 1);
    // fuse + c8
    k_fsum<<<CN / 256, 256, 0, stream>>>(SA, SC, ENC + (size_t)i * 64 * NTOT, FS, CN);
    a2nk(FS, NTOT, HWPIX, 64, 64, 0, true);
    gemm(wbC8[i], BT, FEAT[i], d.c8b, 2, 64, 64, 0);
  }

  // ---- iAFF fusion ----
  float *XA = Q, *Lb = Kb, *Gb = V, *XI = OA;             // reuse head scratch
  auto branch = [&](const float* in, const unsigned short* w1b,
                    const unsigned short* w2b, const Br& p, float* out) {
    a2nk(in, NTOT, HWPIX, 64, 64, 0, true);
    gemm(w1b, BT, Hb, p.b1, 1, 64, 16, 0);
    bn(Hb, Hb, p.g1, p.bt1, 16, 1);
    a2nk(Hb, NTOT, HWPIX, 16, 32, 0, true);
    gemm(w2b, BT, out, p.b2, 2, 32, 64, 0);
    bn(out, out, p.g2, p.bt2, 64, 0);
  };
  auto iaff = [&](const float* fx, const float* fr, float* out) {
    k_add<<<CN / 256, 256, 0, stream>>>(fx, fr, XA, CN);
    branch(XA, wbI1[0], wbI2[0], brL, Lb);
    branch(XA, wbI1[1], wbI2[1], brG, Gb);
    k_mix<<<CN / 256, 256, 0, stream>>>(fx, fr, Lb, Gb, XI, CN);
    branch(XI, wbI1[2], wbI2[2], brL2, Lb);
    branch(XI, wbI1[1], wbI2[1], brG, Gb);                // source reuses glob
    k_mix<<<CN / 256, 256, 0, stream>>>(fx, fr, Lb, Gb, out, CN);
  };
  iaff(FEAT[0], FEAT[1], A123[0]);
  iaff(FEAT[2], FEAT[3], A123[1]);
  iaff(FEAT[4], FEAT[5], A123[2]);

  // ---- concat -> fin conv1x1 -> + x ----
  const float* srcs[8] = {A123[0], A123[1], A123[2], A123[0], A123[1], A123[2],
                          FEAT[0], FEAT[5]};
  for (int j = 0; j < 8; ++j)
    a2nk(srcs[j], NTOT, HWPIX, 64, 512, j * 64, false);   // partial columns
  gemm(wbFIN, BT, ENC /*as FIN*/, fin_b, 16, 512, 512, 0);
  k_final<<<(2 * 512 * HWPIX) / 256, 256, 0, stream>>>(ENC, x_in, (float*)d_out);
}
